// my_SimCLR3_7241314861697
// MI455X (gfx1250) — matrix operand order repaired and checked by reading, not compiled, not run on hardware
//
#include <hip/hip_runtime.h>
#include <hip/hip_bf16.h>
#include <cstdint>

// ---------------- CDNA5 WMMA types ----------------
typedef __attribute__((ext_vector_type(16))) __bf16 v16bf;
typedef __attribute__((ext_vector_type(8)))  __bf16 v8bf;
typedef __attribute__((ext_vector_type(8)))  float  v8f;

union V16 { v16bf v; v8bf h[2]; };

// ---------------- problem constants ----------------
#define B_ROWS   2048
#define KSAMP    64
#define A_DIM    312
#define A_PAD    320      // K padded to multiple of 32 for WMMA
#define P_DIM    128
#define H1_DIM   1024
#define H2_DIM   2048
#define CHUNK    256      // h2 chunk width for fused layer2+3
#define INV_T    (1.0f / 0.12f)

// ---------------- workspace layout (bytes) ----------------
#define ATT_OFF   0u                      // [2048][320] f32
#define QRY_OFF   2621440u                // [2048][128] f32
#define WB1A_OFF  3670016u                // [320][1024] bf16 (K-padded W1a)
#define WB1B_OFF  4325376u                // [1024][2048] bf16
#define WB1C_OFF  8519680u                // [2048][128] bf16
#define WB2B_OFF  9043968u                // [1024][2048] bf16
#define WB2C_OFF  13238272u               // [2048][320] bf16 (N-padded W2c)
// total: 14548992 bytes (~14.5 MB)

// ============================================================
// Fragment loads
// ============================================================
// A-frag: 16x32 bf16 tile from LDS, row-major, stride elements.
// lane&15 = M row; lane>=16 selects the K=+8 half; elements [0..7] are
// K = k0+sel*8 .. +7, elements [8..15] are K = k0+16+sel*8 .. +7.
__device__ __forceinline__ v16bf load_a_frag(const __bf16* hin, int stride,
                                             int k0, int lane) {
    const int m   = lane & 15;
    const int sel = lane >> 4;
    const __bf16* p = hin + m * stride + k0 + sel * 8;
    V16 f;
    f.h[0] = *(const v8bf*)(p);
    f.h[1] = *(const v8bf*)(p + 16);
    return f.v;
}

// B-frag: 32x16 bf16 tile from global weight, natural row-major [K][N].
// lane&15 = N col; elements [0..7] are K = k0+sel*8 .. +7, [8..15] are K = k0+16+sel*8 .. +7.
__device__ __forceinline__ v16bf load_b_frag(const __bf16* wb, int nstride,
                                             int k0, int n0, int lane) {
    const int sel = lane >> 4;
    const __bf16* p = wb + (size_t)(k0 + sel * 8) * nstride + n0 + (lane & 15);
    v16bf f;
#pragma unroll
    for (int s = 0; s < 8; ++s) {
        f[s]     = p[(size_t)s * nstride];
        f[s + 8] = p[(size_t)(s + 16) * nstride];
    }
    return f;
}

// Epilogue: bias + relu + bf16 store of one 16x16 C tile into LDS/global.
__device__ __forceinline__ void store_relu_bf16(v8f c, float bs, __bf16* out,
                                                int stride, int n, int lane) {
    const int mbase = (lane >> 4) << 3;   // C layout: vgpr r -> M = r + 8*sel
#pragma unroll
    for (int r = 0; r < 8; ++r) {
        float v = c[r] + bs;
        v = v > 0.0f ? v : 0.0f;
        out[(mbase + r) * stride + n] = (__bf16)v;
    }
}

// ============================================================
// Register-blocked GEMM micro-kernel: one B fragment feeds 4 M-tiles
// (64 rows). A from LDS [64][astride] bf16, B from global row-major
// weight at row offset wrow0, cols n0..n0+15.
// ============================================================
__device__ __forceinline__ void gemm_4m(const __bf16* ain, int astride,
                                        const __bf16* wb, int nstride,
                                        int wrow0, int n0, int klen,
                                        int lane, v8f c[4]) {
    const __bf16* wrow = wb + (size_t)wrow0 * nstride;
    for (int k0 = 0; k0 < klen; k0 += 32) {
        const v16bf bfr = load_b_frag(wrow, nstride, k0, n0, lane);
#pragma unroll
        for (int mt = 0; mt < 4; ++mt) {
            const v16bf a = load_a_frag(ain + (size_t)(mt * 16) * astride,
                                        astride, k0, lane);
            c[mt] = __builtin_amdgcn_wmma_f32_16x16x32_bf16(
                        false, a, false, bfr, (short)0, c[mt], false, false);
        }
    }
}

// ============================================================
// Simple 16-row layer (used by the small att/query kernels):
//   out = relu(hin[16xkpad] @ W[kpad x nout] + bias)
// ============================================================
__device__ __forceinline__ void mlp_layer(const __bf16* hin, int kpad,
                                          const __bf16* wb,
                                          const float* bias, int nbias,
                                          int nout,
                                          __bf16* out_bf, float* out_f32,
                                          int out_stride, int tid) {
    const int lane   = tid & 31;
    const int wave   = tid >> 5;
    const int ntiles = nout >> 4;
    for (int nt = wave; nt < ntiles; nt += 8) {
        const int n0 = nt << 4;
        v8f c = {};
        for (int k0 = 0; k0 < kpad; k0 += 32) {
            v16bf a = load_a_frag(hin, kpad, k0, lane);
            v16bf b = load_b_frag(wb, nout, k0, n0, lane);
            c = __builtin_amdgcn_wmma_f32_16x16x32_bf16(
                    false, a, false, b, (short)0, c, false, false);
        }
        const int n     = n0 + (lane & 15);
        const float bs  = (n < nbias) ? bias[n] : 0.0f;
        const int mbase = (lane >> 4) << 3;
#pragma unroll
        for (int r = 0; r < 8; ++r) {
            float v = c[r] + bs;
            v = v > 0.0f ? v : 0.0f;
            const int m = mbase + r;
            if (out_bf)  out_bf[m * out_stride + n]  = (__bf16)v;
            if (out_f32) out_f32[m * out_stride + n] = v;
        }
    }
}

// ============================================================
// Kernel 0: fp32 weight [K][N] -> bf16 [Kpad][Npad], zero padded.
// ============================================================
__global__ void __launch_bounds__(256)
cvt_weight_bf16_kernel(const float* __restrict__ W, __bf16* __restrict__ Wb,
                       int K, int N, int Kpad, int Npad) {
    const long long i = (long long)blockIdx.x * 256 + threadIdx.x;
    const long long total = (long long)Kpad * Npad;
    if (i >= total) return;
    const int k = (int)(i / Npad);
    const int n = (int)(i % Npad);
    float v = (k < K && n < N) ? W[(long long)k * N + n] : 0.0f;
    Wb[i] = (__bf16)v;
}

// ============================================================
// Kernel 1: att = MLP2(tar_atts) : 1 -> 1024 -> 2048 -> 312(pad 320)
// One block = 16 rows. Layer 1 is an outer product (elementwise).
// ============================================================
__global__ void __launch_bounds__(256)
att_mlp_kernel(const float* __restrict__ tar,
               const float* __restrict__ w2a, const float* __restrict__ b2a,
               const __bf16* __restrict__ wb2b, const float* __restrict__ b2b,
               const __bf16* __restrict__ wb2c, const float* __restrict__ b2c,
               float* __restrict__ att) {
    extern __shared__ char smem[];
    __bf16* h1 = (__bf16*)(smem);
    __bf16* h2 = (__bf16*)(smem + 16 * H1_DIM * 2);
    const int tid  = threadIdx.x;
    const int row0 = blockIdx.x * 16;

    for (int t = tid; t < 16 * H1_DIM; t += 256) {
        const int m = t >> 10, j = t & (H1_DIM - 1);
        float v = tar[row0 + m] * w2a[j] + b2a[j];
        h1[t] = (__bf16)(v > 0.0f ? v : 0.0f);
    }
    __syncthreads();
    mlp_layer(h1, H1_DIM, wb2b, b2b, H2_DIM, H2_DIM, h2, nullptr, H2_DIM, tid);
    __syncthreads();
    mlp_layer(h2, H2_DIM, wb2c, b2c, A_DIM, A_PAD, nullptr,
              att + (size_t)row0 * A_PAD, A_PAD, tid);
}

// ============================================================
// Kernel 2: query = MLP1(v2s + att) : 312(320) -> 1024 -> 2048 -> 128
// One block = 16 rows.
// ============================================================
__global__ void __launch_bounds__(256)
query_mlp_kernel(const float* __restrict__ v2s, const float* __restrict__ att,
                 const __bf16* __restrict__ wb1a, const float* __restrict__ b1a,
                 const __bf16* __restrict__ wb1b, const float* __restrict__ b1b,
                 const __bf16* __restrict__ wb1c, const float* __restrict__ b1c,
                 float* __restrict__ query) {
    extern __shared__ char smem[];
    __bf16* xs = (__bf16*)(smem);
    __bf16* h1 = (__bf16*)(smem + 16 * A_PAD * 2);
    __bf16* h2 = (__bf16*)(smem + 16 * A_PAD * 2 + 16 * H1_DIM * 2);
    const int tid  = threadIdx.x;
    const int row0 = blockIdx.x * 16;

    for (int t = tid; t < 16 * A_PAD; t += 256) {
        const int m = t / A_PAD, c = t % A_PAD;
        float v = 0.0f;
        if (c < A_DIM)
            v = v2s[(size_t)(row0 + m) * A_DIM + c] + att[(size_t)(row0 + m) * A_PAD + c];
        xs[t] = (__bf16)v;
    }
    __syncthreads();
    mlp_layer(xs, A_PAD, wb1a, b1a, H1_DIM, H1_DIM, h1, nullptr, H1_DIM, tid);
    __syncthreads();
    mlp_layer(h1, H1_DIM, wb1b, b1b, H2_DIM, H2_DIM, h2, nullptr, H2_DIM, tid);
    __syncthreads();
    mlp_layer(h2, H2_DIM, wb1c, b1c, P_DIM, P_DIM, nullptr,
              query + (size_t)row0 * P_DIM, P_DIM, tid);
}

// ============================================================
// Kernel 3: big fused kernel. One block = one b = 64 (b,k) rows.
//   x = v2s[idx[b,:]] + att[b]; p = MLP1(x); logits = dot(query[b], p)/T
// Layer 2+3 fused chunk-wise so h2 (64x2048) is never materialized:
//   for each 256-col chunk of h2: compute chunk in LDS (WMMA), then
//   accumulate chunk @ W1c into per-wave register accumulators.
// LDS: x 40KB + h1 128KB + h2chunk 32KB + p 32KB + q 0.5KB = 232.5KB.
// 2048 blocks x 256 threads (8 waves).
// ============================================================
__global__ void __launch_bounds__(256)
fused_sample_mlp_kernel(const float* __restrict__ v2s,
                        const int* __restrict__ sidx,
                        const float* __restrict__ att,
                        const float* __restrict__ query,
                        const __bf16* __restrict__ wb1a, const float* __restrict__ b1a,
                        const __bf16* __restrict__ wb1b, const float* __restrict__ b1b,
                        const __bf16* __restrict__ wb1c, const float* __restrict__ b1c,
                        float* __restrict__ logits) {
    extern __shared__ char smem[];
    __bf16* xs  = (__bf16*)(smem);                               // 64 x 320
    __bf16* h1  = (__bf16*)(smem + 40960);                       // 64 x 1024
    __bf16* h2c = (__bf16*)(smem + 40960 + 131072);              // 64 x 256
    float*  ps  = (float*) (smem + 40960 + 131072 + 32768);      // 64 x 128
    float*  qs  = (float*) (smem + 40960 + 131072 + 32768 + 32768); // 128

    const int tid  = threadIdx.x;
    const int lane = tid & 31;
    const int wave = tid >> 5;
    const int b    = blockIdx.x;          // one block per batch row
    const int row0 = b * KSAMP;           // flat (b,k) row base

    if (tid < P_DIM) qs[tid] = query[(size_t)b * P_DIM + tid];

    // stage x = v2s[sample] + att[b], bf16, K-padded to 320
    for (int t = tid; t < KSAMP * A_PAD; t += 256) {
        const int m = t / A_PAD, cc = t % A_PAD;
        float v = 0.0f;
        if (cc < A_DIM) {
            const int src = sidx[row0 + m];
            v = v2s[(size_t)src * A_DIM + cc] + att[(size_t)b * A_PAD + cc];
        }
        xs[t] = (__bf16)v;
    }
    __syncthreads();

    // ---- layer 1: h1 = relu(x @ W1a + b1a), 64 ntiles over 8 waves ----
    for (int nt = wave; nt < (H1_DIM >> 4); nt += 8) {
        const int n0 = nt << 4;
        v8f c[4] = {};
        gemm_4m(xs, A_PAD, wb1a, H1_DIM, 0, n0, A_PAD, lane, c);
        const int n    = n0 + (lane & 15);
        const float bs = b1a[n];
#pragma unroll
        for (int mt = 0; mt < 4; ++mt)
            store_relu_bf16(c[mt], bs, h1 + (size_t)(mt * 16) * H1_DIM,
                            H1_DIM, n, lane);
    }
    __syncthreads();

    // ---- layers 2+3 fused, chunked over h2 columns ----
    v8f acc[4] = {};                      // layer-3 accum: wave owns ntile=wave
    const int n3 = (wave << 4) + (lane & 15);

    for (int ch = 0; ch < (H2_DIM / CHUNK); ++ch) {
        // layer 2 chunk: h2c = relu(h1 @ W1b[:, ch*256 .. +256) + b1b)
        for (int nt = wave; nt < (CHUNK >> 4); nt += 8) {
            const int ng0 = ch * CHUNK + (nt << 4);    // global h2 col
            v8f c[4] = {};
            gemm_4m(h1, H1_DIM, wb1b, H2_DIM, 0, ng0, H1_DIM, lane, c);
            const int nl   = (nt << 4) + (lane & 15);  // local chunk col
            const float bs = b1b[ch * CHUNK + nl];
#pragma unroll
            for (int mt = 0; mt < 4; ++mt)
                store_relu_bf16(c[mt], bs, h2c + (size_t)(mt * 16) * CHUNK,
                                CHUNK, nl, lane);
        }
        __syncthreads();
        // layer 3 partial: acc += h2c @ W1c[ch*256 .. +256, wave*16 .. +16)
        gemm_4m(h2c, CHUNK, wb1c, P_DIM, ch * CHUNK, wave << 4, CHUNK, lane, acc);
        __syncthreads();                  // before next chunk overwrites h2c
    }

    // ---- layer 3 epilogue: bias + relu -> ps (LDS f32 [64][128]) ----
    {
        const float bs  = b1c[n3];
        const int mbase = (lane >> 4) << 3;
#pragma unroll
        for (int mt = 0; mt < 4; ++mt) {
#pragma unroll
            for (int r = 0; r < 8; ++r) {
                float v = acc[mt][r] + bs;
                v = v > 0.0f ? v : 0.0f;
                ps[(mt * 16 + mbase + r) * P_DIM + n3] = v;
            }
        }
    }
    __syncthreads();

    // ---- logits[row] = dot(query[b], p[row]) / T ----
    if (tid < KSAMP) {
        float a = 0.0f;
#pragma unroll 4
        for (int c = 0; c < P_DIM; ++c) a += qs[c] * ps[tid * P_DIM + c];
        logits[row0 + tid] = a * INV_T;
    }
}

// ============================================================
// Host launcher
// ============================================================
extern "C" void kernel_launch(void* const* d_in, const int* in_sizes, int n_in,
                              void* d_out, int out_size, void* d_ws, size_t ws_size,
                              hipStream_t stream) {
    (void)in_sizes; (void)n_in; (void)out_size; (void)ws_size;

    const float* v2s   = (const float*)d_in[0];
    const float* tar   = (const float*)d_in[1];
    const int*   sidx  = (const int*)  d_in[2];
    const float* W2a   = (const float*)d_in[3];
    const float* b2a   = (const float*)d_in[4];
    const float* W2b   = (const float*)d_in[5];
    const float* b2b   = (const float*)d_in[6];
    const float* W2c   = (const float*)d_in[7];
    const float* b2c   = (const float*)d_in[8];
    const float* W1a   = (const float*)d_in[9];
    const float* b1a   = (const float*)d_in[10];
    const float* W1b   = (const float*)d_in[11];
    const float* b1b   = (const float*)d_in[12];
    const float* W1c   = (const float*)d_in[13];
    const float* b1c   = (const float*)d_in[14];
    float* logits = (float*)d_out;

    char* ws = (char*)d_ws;
    float*  att  = (float*) (ws + ATT_OFF);
    float*  qry  = (float*) (ws + QRY_OFF);
    __bf16* wb1a = (__bf16*)(ws + WB1A_OFF);
    __bf16* wb1b = (__bf16*)(ws + WB1B_OFF);
    __bf16* wb1c = (__bf16*)(ws + WB1C_OFF);
    __bf16* wb2b = (__bf16*)(ws + WB2B_OFF);
    __bf16* wb2c = (__bf16*)(ws + WB2C_OFF);

    // --- convert weights to bf16 (padded, row-major [Kpad][Npad]) ---
    auto cvt = [&](const float* W, __bf16* Wb, int K, int N, int Kp, int Np) {
        long long total = (long long)Kp * Np;
        int blocks = (int)((total + 255) / 256);
        cvt_weight_bf16_kernel<<<blocks, 256, 0, stream>>>(W, Wb, K, N, Kp, Np);
    };
    cvt(W1a, wb1a, A_DIM,  H1_DIM, A_PAD,  H1_DIM);
    cvt(W1b, wb1b, H1_DIM, H2_DIM, H1_DIM, H2_DIM);
    cvt(W1c, wb1c, H2_DIM, P_DIM,  H2_DIM, P_DIM);
    cvt(W2b, wb2b, H1_DIM, H2_DIM, H1_DIM, H2_DIM);
    cvt(W2c, wb2c, H2_DIM, A_DIM,  H2_DIM, A_PAD);

    // --- dynamic LDS sizes ---
    const int att_lds   = 16 * H1_DIM * 2 + 16 * H2_DIM * 2;                 //  96 KB
    const int query_lds = 16 * A_PAD * 2 + att_lds;                          // 106 KB
    const int main_lds  = KSAMP * A_PAD * 2 + KSAMP * H1_DIM * 2 +
                          KSAMP * CHUNK * 2 + KSAMP * P_DIM * 4 + P_DIM * 4; // 232.5 KB
    hipFuncSetAttribute((const void*)att_mlp_kernel,
                        hipFuncAttributeMaxDynamicSharedMemorySize, att_lds);
    hipFuncSetAttribute((const void*)query_mlp_kernel,
                        hipFuncAttributeMaxDynamicSharedMemorySize, query_lds);
    hipFuncSetAttribute((const void*)fused_sample_mlp_kernel,
                        hipFuncAttributeMaxDynamicSharedMemorySize, main_lds);

    // --- att embedding: 2048 rows / 16 = 128 blocks ---
    att_mlp_kernel<<<B_ROWS / 16, 256, att_lds, stream>>>(
        tar, W2a, b2a, wb2b, b2b, wb2c, b2c, att);

    // --- query projection: 128 blocks ---
    query_mlp_kernel<<<B_ROWS / 16, 256, query_lds, stream>>>(
        v2s, att, wb1a, b1a, wb1b, b1b, wb1c, b1c, qry);

    // --- fused per-sample MLP + logits: one block per b = 2048 blocks ---
    fused_sample_mlp_kernel<<<B_ROWS, 256, main_lds, stream>>>(
        v2s, sidx, att, qry, wb1a, b1a, wb1b, b1b, wb1c, b1c, logits);
}
